// AlignmentWithProtoype_54546084659724
// MI455X (gfx1250) — matrix operand order repaired and checked by reading, not compiled, not run on hardware
//
#include <hip/hip_runtime.h>
#include <hip/hip_bf16.h>

typedef __attribute__((ext_vector_type(16))) __bf16 v16bf;
typedef __attribute__((ext_vector_type(8)))  float  v8f;

#if defined(__HIP_DEVICE_COMPILE__) && __has_builtin(__builtin_amdgcn_tensor_load_to_lds)
#define AWP_HAS_TDM 1
typedef __attribute__((ext_vector_type(4))) unsigned int u32x4;
typedef __attribute__((ext_vector_type(8))) int          i32x8;
typedef __attribute__((ext_vector_type(4))) int          i32x4;
#endif

#define NROWS   131072      // 128*1024 tokens
#define DDIM    192
#define MPROT   128
#define KPROT   256         // 2*M prototypes
#define INV_T   20.0f       // 1/0.05
#define DECAY   0.96f
#define CSBLOCKS 1024       // blocks in the fused sinkhorn pass (128 rows each)

// ---------------------------------------------------------------------------
// K0: normalize concat(local, global) prototypes, emit bf16 [256][192]
// ---------------------------------------------------------------------------
__global__ void awp_protos(const float* __restrict__ lp, const float* __restrict__ gp,
                           __bf16* __restrict__ pn) {
  int p = threadIdx.x;                       // 0..255
  const float* src = (p < MPROT) ? (lp + p * DDIM) : (gp + (p - MPROT) * DDIM);
  float ss = 0.f;
  for (int k = 0; k < DDIM; ++k) { float t = src[k]; ss += t * t; }
  float rn = rsqrtf(fmaxf(ss, 1e-12f));
  __bf16* dst = pn + p * DDIM;
  for (int k = 0; k < DDIM; ++k) dst[k] = (__bf16)(src[k] * rn);
}

// ---------------------------------------------------------------------------
// K1: Q[i][j] = exp( (l2norm(x_i) . pn_j) / T )   via bf16 WMMA
//   block = 256 threads (8 waves); wave -> 16 rows; block -> 128 rows
//   LDS: bf16 prototypes [256][192] (98304 B, dynamic), staged by the TDM
// ---------------------------------------------------------------------------
__global__ void awp_gemm(const float* __restrict__ x, const unsigned* __restrict__ pnb,
                         float* __restrict__ Q) {
  extern __shared__ unsigned smem_u[];

#ifdef AWP_HAS_TDM
  // Tensor Data Mover: one wave DMAs the 98304-byte prototype block into LDS.
  if (threadIdx.x < 32) {
    unsigned long long ga = (unsigned long long)(__SIZE_TYPE__)pnb;
    unsigned lds_off = (unsigned)(__SIZE_TYPE__)(void*)smem_u; // low 32b = LDS byte offset
    u32x4 g0;
    g0[0] = 1u;                                            // count=1, user mode
    g0[1] = lds_off;                                       // lds_addr
    g0[2] = (unsigned)(ga & 0xFFFFFFFFu);                  // global_addr[31:0]
    g0[3] = (unsigned)((ga >> 32) & 0x01FFFFFFu) | (2u << 30); // addr[56:32] | type=2
    i32x8 g1;
    g1[0] = 0x00020000;   // workgroup_mask=0, data_size=2 (4 bytes)
    g1[1] = 0x60000000;   // tensor_dim0[15:0]=24576 in bits[63:48]
    g1[2] = 0x00010000;   // tensor_dim0 hi=0, tensor_dim1=1
    g1[3] = 0x60000000;   // tensor_dim1 hi=0, tile_dim0=24576
    g1[4] = 0x00000001;   // tile_dim1=1, tile_dim2=0
    g1[5] = 24576;        // tensor_dim0_stride low
    g1[6] = 0;
    g1[7] = 0;
    i32x4 gz4 = {0, 0, 0, 0};
    i32x8 gz8 = {0, 0, 0, 0, 0, 0, 0, 0};
    __builtin_amdgcn_tensor_load_to_lds(g0, g1, gz4, gz4, gz8, 0);
    __builtin_amdgcn_s_wait_tensorcnt((short)0);
  }
#else
  for (int i = threadIdx.x; i < (KPROT * DDIM / 2); i += 256) smem_u[i] = pnb[i];
#endif
  __syncthreads();
  const __bf16* sprot = (const __bf16*)smem_u;

  const int wave = threadIdx.x >> 5;
  const int lane = threadIdx.x & 31;
  const int lrow = lane & 15;
  const int ksel = lane >> 4;                       // which half of the wave
  const long r0  = (long)blockIdx.x * 128 + wave * 16;
  const float* xr = x + (r0 + lrow) * DDIM;

  // row sum-of-squares: lane pair (L, L+16) splits the 192-dim row
  float ss = 0.f;
  {
    const float4* p4 = (const float4*)(xr + 96 * ksel);
#pragma unroll
    for (int t = 0; t < 24; ++t) {
      float4 q4 = p4[t];
      ss += q4.x * q4.x + q4.y * q4.y + q4.z * q4.z + q4.w * q4.w;
    }
  }
  ss += __shfl_xor(ss, 16, 32);
  const float rn = rsqrtf(fmaxf(ss, 1e-12f));

  // A fragments for all 6 K-steps (contiguous K order, matching B):
  // lanes 0-15: K {k0..k0+15}; lanes 16-31: K {k0+16..k0+31}
  v16bf a[6];
#pragma unroll
  for (int kk = 0; kk < 6; ++kk) {
    const float* pa = xr + kk * 32 + 16 * ksel;
    float4 f0 = *(const float4*)(pa);
    float4 f1 = *(const float4*)(pa + 4);
    float4 f2 = *(const float4*)(pa + 8);
    float4 f3 = *(const float4*)(pa + 12);
    a[kk][0]  = (__bf16)(f0.x * rn); a[kk][1]  = (__bf16)(f0.y * rn);
    a[kk][2]  = (__bf16)(f0.z * rn); a[kk][3]  = (__bf16)(f0.w * rn);
    a[kk][4]  = (__bf16)(f1.x * rn); a[kk][5]  = (__bf16)(f1.y * rn);
    a[kk][6]  = (__bf16)(f1.z * rn); a[kk][7]  = (__bf16)(f1.w * rn);
    a[kk][8]  = (__bf16)(f2.x * rn); a[kk][9]  = (__bf16)(f2.y * rn);
    a[kk][10] = (__bf16)(f2.z * rn); a[kk][11] = (__bf16)(f2.w * rn);
    a[kk][12] = (__bf16)(f3.x * rn); a[kk][13] = (__bf16)(f3.y * rn);
    a[kk][14] = (__bf16)(f3.z * rn); a[kk][15] = (__bf16)(f3.w * rn);
  }

  // sweep 16 column tiles; B frag: lane n = proto col c0+n, contiguous K run
  for (int ct = 0; ct < 16; ++ct) {
    const int c0 = ct * 16;
    v8f acc = {0.f, 0.f, 0.f, 0.f, 0.f, 0.f, 0.f, 0.f};
#pragma unroll
    for (int kk = 0; kk < 6; ++kk) {
      const v16bf b = *(const v16bf*)(sprot + (c0 + lrow) * DDIM + kk * 32 + 16 * ksel);
      acc = __builtin_amdgcn_wmma_f32_16x16x32_bf16(false, a[kk], false, b,
                                                    (short)0, acc, false, false);
    }
    // D layout: VGPR r -> row r0 + r + 8*ksel, col c0 + lrow
    float* qb = Q + (r0 + 8 * ksel) * KPROT + c0 + lrow;
#pragma unroll
    for (int r = 0; r < 8; ++r)
      qb[(long)r * KPROT] = __expf(acc[r] * INV_T);
  }
}

// ---------------------------------------------------------------------------
// Sinkhorn (scale-reduced): Q = q * u_i * v_j;  u_i = 1/<q_i,v>, v_j = 1/sum_i q_ij u_i
// Fused pass: one Q read per iteration computes u on the fly and accumulates
// per-block column partials (deterministic two-stage reduction, no atomics).
// ---------------------------------------------------------------------------
__global__ void awp_vinit(float* v) { v[threadIdx.x] = 1.0f; }

__global__ void awp_iter(const float* __restrict__ Q, const float* __restrict__ v,
                         float* __restrict__ partial) {
  __shared__ float sacc[8 * KPROT];
  const int wave = threadIdx.x >> 5, lane = threadIdx.x & 31;
  const float4 v0 = *(const float4*)(v + lane * 4);
  const float4 v1 = *(const float4*)(v + 128 + lane * 4);
  float4 a0 = {0.f, 0.f, 0.f, 0.f}, a1 = {0.f, 0.f, 0.f, 0.f};
  const long base = (long)blockIdx.x * 128 + wave * 16;
  for (int r = 0; r < 16; ++r) {
    const float* q = Q + (base + r) * KPROT;
    __builtin_prefetch(q + 2 * KPROT, 0, 1);      // global_prefetch_b8 two rows ahead
    float4 q0 = *(const float4*)(q + lane * 4);
    float4 q1 = *(const float4*)(q + 128 + lane * 4);
    float s = q0.x * v0.x + q0.y * v0.y + q0.z * v0.z + q0.w * v0.w
            + q1.x * v1.x + q1.y * v1.y + q1.z * v1.z + q1.w * v1.w;
    for (int off = 16; off; off >>= 1) s += __shfl_xor(s, off, 32); // all lanes get sum
    const float uu = 1.0f / s;
    a0.x += q0.x * uu; a0.y += q0.y * uu; a0.z += q0.z * uu; a0.w += q0.w * uu;
    a1.x += q1.x * uu; a1.y += q1.y * uu; a1.z += q1.z * uu; a1.w += q1.w * uu;
  }
  float* sl = sacc + wave * KPROT;
  ((float4*)sl)[lane]         = a0;   // cols 4l..4l+3
  ((float4*)(sl + 128))[lane] = a1;   // cols 128+4l..128+4l+3
  __syncthreads();
  const int col = threadIdx.x;        // 0..255
  float p = 0.f;
#pragma unroll
  for (int w2 = 0; w2 < 8; ++w2) p += sacc[w2 * KPROT + col];
  partial[(long)blockIdx.x * KPROT + col] = p;
}

__global__ void awp_vfin(const float* __restrict__ partial, float* __restrict__ v) {
  float s = 0.f;
  for (int b = 0; b < CSBLOCKS; ++b) s += partial[(long)b * KPROT + threadIdx.x];
  v[threadIdx.x] = 1.0f / s;
}

// ---------------------------------------------------------------------------
// K3: per-row argmax over q_ij * v_j (local half / global half)
// ---------------------------------------------------------------------------
__global__ void awp_mrinit(int* maxrow) { maxrow[threadIdx.x] = -1; }

__global__ void awp_assign(const float* __restrict__ Q, const float* __restrict__ v,
                           int* __restrict__ gassign, int* __restrict__ maxrow) {
  int wave = threadIdx.x >> 5, lane = threadIdx.x & 31;
  long i = (long)blockIdx.x * 8 + wave;
  const float* q = Q + i * KPROT;
  float bl = -1e30f, bg = -1e30f; int bli = 0, bgi = 0;
#pragma unroll
  for (int t = 0; t < 4; ++t) {
    int j = lane * 4 + t;
    float vl = q[j] * v[j];
    if (vl > bl) { bl = vl; bli = j; }
    float vg = q[128 + j] * v[128 + j];
    if (vg > bg) { bg = vg; bgi = j; }
  }
  for (int off = 16; off; off >>= 1) {
    float ol = __shfl_xor(bl, off, 32); int oli = __shfl_xor(bli, off, 32);
    if (ol > bl || (ol == bl && oli < bli)) { bl = ol; bli = oli; }
    float og = __shfl_xor(bg, off, 32); int ogi = __shfl_xor(bgi, off, 32);
    if (og > bg || (og == bg && ogi < bgi)) { bg = og; bgi = ogi; }
  }
  if (lane == 0) {
    gassign[i] = bgi;
    atomicMax(&maxrow[bli], (int)i);
  }
}

// ---------------------------------------------------------------------------
// K4: gated GLU table over the 128 global prototypes (tiny GEMM)
// ---------------------------------------------------------------------------
__global__ void awp_glu(const float* __restrict__ gp, const float* __restrict__ W,
                        const float* __restrict__ b, float* __restrict__ gated) {
  __shared__ float sg[DDIM];
  int m = blockIdx.x, d = threadIdx.x;
  sg[d] = gp[m * DDIM + d];
  __syncthreads();
  float l1 = b[d], l2 = b[DDIM + d];
  for (int k = 0; k < DDIM; ++k) {
    float g = sg[k];
    l1 += g * W[k * 2 * DDIM + d];
    l2 += g * W[k * 2 * DDIM + DDIM + d];
  }
  gated[m * DDIM + d] = l1 * (1.0f / (1.0f + __expf(-l2)));
}

// ---------------------------------------------------------------------------
// K5: out = l2norm(0.5*gated[g_i] + 0.5*x_i)  (one wave per row)
// ---------------------------------------------------------------------------
__global__ void awp_out(const float* __restrict__ x, const float* __restrict__ gated,
                        const int* __restrict__ gassign, float* __restrict__ out) {
  int wave = threadIdx.x >> 5, lane = threadIdx.x & 31;
  long i = (long)blockIdx.x * 8 + wave;
  int g = gassign[i];
  const float* xr = x + i * DDIM;
  const float* gr = gated + (long)g * DDIM;
  float vals[6]; float ss = 0.f;
#pragma unroll
  for (int t = 0; t < 6; ++t) {
    int e = lane + 32 * t;
    float vv = 0.5f * gr[e] + 0.5f * xr[e];
    vals[t] = vv; ss += vv * vv;
  }
  for (int off = 16; off; off >>= 1) ss += __shfl_xor(ss, off, 32);
  float rn = rsqrtf(fmaxf(ss, 1e-12f));
  float* orow = out + i * DDIM;
#pragma unroll
  for (int t = 0; t < 6; ++t) orow[lane + 32 * t] = vals[t] * rn;
}

// ---------------------------------------------------------------------------
// K6: new_local via last-write-wins scatter (maxrow index)
// ---------------------------------------------------------------------------
__global__ void awp_newlocal(const float* __restrict__ lp, const float* __restrict__ x,
                             const int* __restrict__ maxrow, float* __restrict__ outNL) {
  int m = blockIdx.x, d = threadIdx.x;
  int r = maxrow[m];
  float p = lp[m * DDIM + d];
  float val = (r < 0) ? p : (DECAY * p + (1.0f - DECAY) * x[(long)r * DDIM + d]);
  outNL[m * DDIM + d] = val;
}

// ---------------------------------------------------------------------------
extern "C" void kernel_launch(void* const* d_in, const int* in_sizes, int n_in,
                              void* d_out, int out_size, void* d_ws, size_t ws_size,
                              hipStream_t stream) {
  const float* x  = (const float*)d_in[0];   // [N, 192]
  const float* lp = (const float*)d_in[1];   // [128, 192]
  const float* gp = (const float*)d_in[2];   // [128, 192]
  const float* W  = (const float*)d_in[3];   // [192, 384]
  const float* bb = (const float*)d_in[4];   // [384]

  char* w = (char*)d_ws;
  float*    Q       = (float*)w;    w += (size_t)NROWS * KPROT * 4;      // 134 MB
  unsigned* pnb     = (unsigned*)w; w += (size_t)KPROT * DDIM * 2;       // bf16 protos
  float*    v       = (float*)w;    w += 256 * 4;
  float*    partial = (float*)w;    w += (size_t)CSBLOCKS * KPROT * 4;   // 1 MB
  int*      gassign = (int*)w;      w += (size_t)NROWS * 4;
  int*      maxrow  = (int*)w;      w += 128 * 4;
  float*    gated   = (float*)w;    w += (size_t)MPROT * DDIM * 4;

  float* out_normed = (float*)d_out;                        // [N, 192]
  float* out_local  = (float*)d_out + (size_t)NROWS * DDIM; // [128, 192]

  awp_protos<<<1, 256, 0, stream>>>(lp, gp, (__bf16*)pnb);
  awp_gemm<<<NROWS / 128, 256, KPROT * DDIM * 2, stream>>>(x, pnb, Q);

  awp_vinit<<<1, 256, 0, stream>>>(v);
  for (int it = 0; it < 3; ++it) {
    awp_iter<<<CSBLOCKS, 256, 0, stream>>>(Q, v, partial);
    awp_vfin<<<1, 256, 0, stream>>>(partial, v);
  }

  awp_mrinit<<<1, 128, 0, stream>>>(maxrow);
  awp_assign<<<NROWS / 8, 256, 0, stream>>>(Q, v, gassign, maxrow);

  awp_glu<<<MPROT, DDIM, 0, stream>>>(gp, W, bb, gated);
  awp_out<<<NROWS / 8, 256, 0, stream>>>(x, gated, gassign, out_normed);
  awp_newlocal<<<MPROT, DDIM, 0, stream>>>(lp, x, maxrow, out_local);
}
